// AgriSits_64115271794770
// MI455X (gfx1250) — matrix operand order repaired and checked by reading, not compiled, not run on hardware
//
#include <hip/hip_runtime.h>
#include <hip/hip_bf16.h>
#include <math.h>

// ---------------- problem constants ----------------
#define T_ 406
#define D_ 10
#define P_ 128
#define F_ 128
#define B_ 64
#define NCP_ 12
#define ROWS_ (B_ * T_)   // 25984 = 1624 * 16

// padded activation row counts
#define RP1_ 413   // conv1: pad (3,4), Cin padded 10->32
#define RP2_ 410   // conv2: pad (2,2), Cin=128
#define RP3_ 408   // conv3: pad (1,1), Cin=256

typedef __attribute__((ext_vector_type(16))) __bf16 v16bf;
typedef __attribute__((ext_vector_type(8)))  __bf16 v8bf;
typedef __attribute__((ext_vector_type(8)))  float  v8f;
typedef __attribute__((ext_vector_type(8)))  float  v8fl;

__device__ __forceinline__ v8f wmma_bf16(const v16bf& a, const v16bf& b, const v8f& c) {
  return __builtin_amdgcn_wmma_f32_16x16x32_bf16(false, a, false, b, (short)0, c, false, false);
}

// Assemble the 16-bit A fragment from its two contiguous 8-element K-runs
// (ISA 7.12.2: elems 0..7 <-> K=8*hi+0..7, elems 8..15 <-> K=16+8*hi+0..7).
__device__ __forceinline__ v16bf afrag(const __bf16* base, int hi) {
  v8bf lo = *(const v8bf*)(base + 8 * hi);
  v8bf up = *(const v8bf*)(base + 16 + 8 * hi);
  return __builtin_shufflevector(lo, up, 0, 1, 2, 3, 4, 5, 6, 7,
                                 8, 9, 10, 11, 12, 13, 14, 15);
}

// =====================================================================
// TPS constants: build FK (31x31), Gauss-Jordan inverse (partial pivot)
// in LDS, fold into TPS_REP -> M28 (T x 28) and ybase (T).
// =====================================================================
__device__ __forceinline__ float tcpx(int n) { return -1.f + 2.f * (float)(n >> 1) / 13.f; }
__device__ __forceinline__ float tcpy(int n) { return (n & 1) ? 1.f : -1.f; }

__global__ void k_tps(float* M28, float* ybase) {
  __shared__ float aug[31][64];
  __shared__ int piv;
  int tid = threadIdx.x;  // blockDim = 64
  for (int i = tid; i < 31 * 31; i += 64) {
    int r = i / 31, c = i % 31;
    float v = 0.f;
    if (r < 28 && c < 28) {
      float dx = tcpx(r) - tcpx(c), dy = tcpy(r) - tcpy(c);
      float r2 = dx * dx + dy * dy;
      v = (r2 > 0.f) ? 0.5f * r2 * logf(r2) : 0.f;
    } else if (r < 28 && c == 28) v = 1.f;
    else if (r == 28 && c < 28) v = 1.f;
    else if (r < 28 && c == 29) v = tcpx(r);
    else if (r < 28 && c == 30) v = tcpy(r);
    else if (r == 29 && c < 28) v = tcpx(c);
    else if (r == 30 && c < 28) v = tcpy(c);
    aug[r][c] = v;
  }
  for (int i = tid; i < 31 * 31; i += 64) {
    int r = i / 31, c = i % 31;
    aug[r][31 + c] = (r == c) ? 1.f : 0.f;
  }
  __syncthreads();
  for (int k = 0; k < 31; k++) {
    if (tid == 0) {
      int p = k; float best = fabsf(aug[k][k]);
      for (int r = k + 1; r < 31; r++) {
        float v = fabsf(aug[r][k]);
        if (v > best) { best = v; p = r; }
      }
      piv = p;
    }
    __syncthreads();
    int p = piv;
    if (p != k) {
      for (int c = tid; c < 62; c += 64) {
        float tv = aug[k][c]; aug[k][c] = aug[p][c]; aug[p][c] = tv;
      }
    }
    __syncthreads();
    float d = aug[k][k];
    __syncthreads();
    float rd = 1.f / d;
    for (int c = tid; c < 62; c += 64) aug[k][c] *= rd;
    __syncthreads();
    for (int r = tid; r < 31; r += 64) {
      if (r != k) {
        float f = aug[r][k];
        for (int c = k; c < 62; c++) aug[r][c] -= f * aug[k][c];
      }
    }
    __syncthreads();
  }
  for (int t = tid; t < T_; t += 64) {
    float yt = -1.f + 2.f * (float)t / (float)(T_ - 1);
    float rep[31];
    for (int n = 0; n < 28; n++) {
      float dx = -1.f - tcpx(n), dy = yt - tcpy(n);
      float r2 = dx * dx + dy * dy;
      rep[n] = (r2 > 0.f) ? 0.5f * r2 * logf(r2) : 0.f;
    }
    rep[28] = 1.f; rep[29] = -1.f; rep[30] = yt;
    float yb = 0.f;
    for (int m = 0; m < 28; m++) {
      float s = 0.f;
      for (int n = 0; n < 31; n++) s += rep[n] * aug[n][31 + m];
      M28[t * 28 + m] = s;
      yb += s * tcpy(m);
    }
    ybase[t] = yb;
  }
}

// =====================================================================
// Gaussian temporal smoothing (fp32, LDS rows): inp_s, mask_s.
// =====================================================================
__global__ void k_smooth(const float* __restrict__ x, const int* __restrict__ mask,
                         float* __restrict__ inp_s, float* __restrict__ mask_s) {
  int b = blockIdx.x;
  __shared__ float xs[T_ * D_];
  __shared__ float ms[T_];
  for (int i = threadIdx.x; i < T_ * D_; i += 256) xs[i] = x[(size_t)b * T_ * D_ + i];
  for (int i = threadIdx.x; i < T_; i += 256) ms[i] = (float)mask[b * T_ + i];
  __syncthreads();
  for (int t = threadIdx.x; t < T_; t += 256) {
    float acc[D_]; float am = 0.f;
#pragma unroll
    for (int d = 0; d < D_; d++) acc[d] = 0.f;
    for (int i = 0; i < T_; i++) {
      float dt = (float)(t - i);
      float w = __expf(dt * dt * (-1.f / 98.f));
      am += w * ms[i];
#pragma unroll
      for (int d = 0; d < D_; d++) acc[d] += w * xs[i * D_ + d];
    }
    float den = (am == 0.f) ? 1.f : am;
    float rden = 1.f / den;
#pragma unroll
    for (int d = 0; d < D_; d++) inp_s[(size_t)b * T_ * D_ + t * D_ + d] = acc[d] * rden;
    mask_s[b * T_ + t] = am;
  }
}

// =====================================================================
// Activation prep kernels: zero-padded bf16 buffers (B, rowsPad, C).
// =====================================================================
__global__ void k_actx(const float* __restrict__ x, __bf16* __restrict__ xp) {
  // xp: (B, RP1_, 32); interior row tp maps to t = tp-3; channels >=10 zero
  size_t i = (size_t)blockIdx.x * 256 + threadIdx.x;
  if (i >= (size_t)B_ * RP1_ * 32) return;
  int c = (int)(i & 31);
  size_t rp = i >> 5;
  int b = (int)(rp / RP1_);
  int tp = (int)(rp - (size_t)b * RP1_);
  int t = tp - 3;
  float v = 0.f;
  if (c < D_ && t >= 0 && t < T_) v = x[((size_t)b * T_ + t) * D_ + c];
  xp[i] = (__bf16)v;
}

__global__ void k_act(const float* __restrict__ h, const float* __restrict__ aff_a,
                      const float* __restrict__ aff_b, __bf16* __restrict__ act,
                      int C, int padL, int rowsPad) {
  size_t i = (size_t)blockIdx.x * 256 + threadIdx.x;
  if (i >= (size_t)B_ * rowsPad * C) return;
  int c = (int)(i % C);
  size_t rp = i / C;
  int b = (int)(rp / rowsPad);
  int tp = (int)(rp - (size_t)b * rowsPad);
  int t = tp - padL;
  float v = 0.f;
  if (t >= 0 && t < T_)
    v = fmaxf(h[((size_t)b * T_ + t) * C + c] * aff_a[c] + aff_b[c], 0.f);
  act[i] = (__bf16)v;
}

// =====================================================================
// conv1: Cin padded to 32, Kw=8. Per row-tile: 8 WMMAs, pure vector loads.
// =====================================================================
__global__ __launch_bounds__(256) void k_conv1(const __bf16* __restrict__ xp,
                                               const float* __restrict__ w,
                                               const float* __restrict__ bias,
                                               float* __restrict__ h1) {
  const int Cp = 32, Kw = 8, N = 128;
  __shared__ __attribute__((aligned(32))) __bf16 pan[Kw * 16 * Cp];  // 8 KB
  int ct = blockIdx.x;
  for (int i = threadIdx.x; i < Kw * 16 * Cp; i += 256) {
    int kk = i / (16 * Cp), rem = i - kk * 16 * Cp, n = rem >> 5, c = rem & 31;
    float v = (c < D_) ? w[(ct * 16 + n) * 80 + c * 8 + kk] : 0.f;
    pan[i] = (__bf16)v;
  }
  __syncthreads();
  int lane = threadIdx.x & 31, wave = threadIdx.x >> 5;
  int hi = lane >> 4, m = lane & 15, n = lane & 15;
  for (int rt = blockIdx.y * 8 + wave; rt < ROWS_ / 16; rt += gridDim.y * 8) {
    v8f c = {0.f, 0.f, 0.f, 0.f, 0.f, 0.f, 0.f, 0.f};
    int row = rt * 16 + m, b = row / T_, t = row - b * T_;
    const __bf16* ab = xp + ((size_t)b * RP1_ + t) * Cp;
#pragma unroll
    for (int kk = 0; kk < Kw; kk++) {
      v16bf a = afrag(ab + kk * Cp, hi);
      v16bf bb = afrag(&pan[kk * 16 * Cp + n * Cp], hi);
      c = wmma_bf16(a, bb, c);
    }
    float bs = bias[ct * 16 + n];
#pragma unroll
    for (int j = 0; j < 8; j++)
      h1[(size_t)(rt * 16 + j + 8 * hi) * N + ct * 16 + n] = c[j] + bs;
  }
}

// =====================================================================
// conv2: Cin=128, Kw=5. Per row-tile: 20 WMMAs.
// =====================================================================
__global__ __launch_bounds__(256) void k_conv2(const __bf16* __restrict__ act1,
                                               const float* __restrict__ w,
                                               const float* __restrict__ bias,
                                               float* __restrict__ h2) {
  const int Cin = 128, Kw = 5, N = 256;
  __shared__ __attribute__((aligned(32))) __bf16 pan[Kw * 16 * Cin];  // 20 KB
  int ct = blockIdx.x;
  for (int i = threadIdx.x; i < Kw * 16 * Cin; i += 256) {
    int kk = i / (16 * Cin), rem = i - kk * 16 * Cin, n = rem >> 7, c = rem & 127;
    pan[i] = (__bf16)w[(ct * 16 + n) * (Cin * Kw) + c * Kw + kk];
  }
  __syncthreads();
  int lane = threadIdx.x & 31, wave = threadIdx.x >> 5;
  int hi = lane >> 4, m = lane & 15, n = lane & 15;
  for (int rt = blockIdx.y * 8 + wave; rt < ROWS_ / 16; rt += gridDim.y * 8) {
    v8f c = {0.f, 0.f, 0.f, 0.f, 0.f, 0.f, 0.f, 0.f};
    int row = rt * 16 + m, b = row / T_, t = row - b * T_;
    const __bf16* ab = act1 + ((size_t)b * RP2_ + t) * Cin;
#pragma unroll
    for (int kk = 0; kk < Kw; kk++) {
      const __bf16* ar = ab + kk * Cin;
#pragma unroll
      for (int ks = 0; ks < Cin / 32; ks++) {
        v16bf a = afrag(ar + ks * 32, hi);
        v16bf bb = afrag(&pan[kk * 16 * Cin + n * Cin + ks * 32], hi);
        c = wmma_bf16(a, bb, c);
      }
    }
    float bs = bias[ct * 16 + n];
#pragma unroll
    for (int j = 0; j < 8; j++)
      h2[(size_t)(rt * 16 + j + 8 * hi) * N + ct * 16 + n] = c[j] + bs;
  }
}

// =====================================================================
// conv3: Cin=256, Kw=3. Per row-tile: 24 WMMAs.
// =====================================================================
__global__ __launch_bounds__(256) void k_conv3(const __bf16* __restrict__ act2,
                                               const float* __restrict__ w,
                                               const float* __restrict__ bias,
                                               float* __restrict__ h3) {
  const int Cin = 256, Kw = 3, N = 128;
  __shared__ __attribute__((aligned(32))) __bf16 pan[Kw * 16 * Cin];  // 24 KB
  int ct = blockIdx.x;
  for (int i = threadIdx.x; i < Kw * 16 * Cin; i += 256) {
    int kk = i / (16 * Cin), rem = i - kk * 16 * Cin, n = rem >> 8, c = rem & 255;
    pan[i] = (__bf16)w[(ct * 16 + n) * (Cin * Kw) + c * Kw + kk];
  }
  __syncthreads();
  int lane = threadIdx.x & 31, wave = threadIdx.x >> 5;
  int hi = lane >> 4, m = lane & 15, n = lane & 15;
  for (int rt = blockIdx.y * 8 + wave; rt < ROWS_ / 16; rt += gridDim.y * 8) {
    v8f c = {0.f, 0.f, 0.f, 0.f, 0.f, 0.f, 0.f, 0.f};
    int row = rt * 16 + m, b = row / T_, t = row - b * T_;
    const __bf16* ab = act2 + ((size_t)b * RP3_ + t) * Cin;
#pragma unroll
    for (int kk = 0; kk < Kw; kk++) {
      const __bf16* ar = ab + kk * Cin;
#pragma unroll
      for (int ks = 0; ks < Cin / 32; ks++) {
        v16bf a = afrag(ar + ks * 32, hi);
        v16bf bb = afrag(&pan[kk * 16 * Cin + n * Cin + ks * 32], hi);
        c = wmma_bf16(a, bb, c);
      }
    }
    float bs = bias[ct * 16 + n];
#pragma unroll
    for (int j = 0; j < 8; j++)
      h3[(size_t)(rt * 16 + j + 8 * hi) * N + ct * 16 + n] = c[j] + bs;
  }
}

// =====================================================================
// Batch-norm stats -> fused affine (a, b): y = a[c]*x + b[c].
// =====================================================================
__global__ void k_stats(const float* __restrict__ h, int C,
                        const float* __restrict__ g, const float* __restrict__ beta,
                        float* __restrict__ a, float* __restrict__ b2) {
  int c = blockIdx.x;
  float s = 0.f, ss = 0.f;
  for (int r = threadIdx.x; r < ROWS_; r += 256) {
    float v = h[(size_t)r * C + c];
    s += v; ss += v * v;
  }
  __shared__ float sh[256], sh2[256];
  sh[threadIdx.x] = s; sh2[threadIdx.x] = ss;
  __syncthreads();
  for (int o = 128; o > 0; o >>= 1) {
    if (threadIdx.x < o) { sh[threadIdx.x] += sh[threadIdx.x + o]; sh2[threadIdx.x] += sh2[threadIdx.x + o]; }
    __syncthreads();
  }
  if (threadIdx.x == 0) {
    float m = sh[0] / (float)ROWS_;
    float var = sh2[0] / (float)ROWS_ - m * m;
    float av = g[c] * rsqrtf(var + 1e-5f);
    a[c] = av;
    b2[c] = beta[c] - m * av;
  }
}

// =====================================================================
// bn3 + relu + global average pool over time -> pooled (B x 128)
// =====================================================================
__global__ void k_pool(const float* __restrict__ h3, const float* __restrict__ a3,
                       const float* __restrict__ b3, float* __restrict__ pooled) {
  int b = blockIdx.x, c = threadIdx.x;  // 128 threads
  float av = a3[c], bv = b3[c], s = 0.f;
  const float* hb = h3 + (size_t)b * T_ * 128 + c;
  for (int t = 0; t < T_; t++) s += fmaxf(hb[t * 128] * av + bv, 0.f);
  pooled[b * 128 + c] = s * (1.f / (float)T_);
}

// =====================================================================
// WMMA GEMM: A (64 x 128) fp32, B (128 x N) fp32 row-major, bias + act.
// act: 0 none, 1 tanh*5, 2 tanh.
// =====================================================================
__global__ __launch_bounds__(256) void k_gemm128(const float* __restrict__ A,
                                                 const float* __restrict__ Bm,
                                                 const float* __restrict__ bias,
                                                 float* __restrict__ Cm, int N, int act) {
  __shared__ __attribute__((aligned(32))) __bf16 pan[16 * 128];
  int ct = blockIdx.x;
  for (int i = threadIdx.x; i < 16 * 128; i += 256) {
    int n = i >> 7, k = i & 127;
    pan[i] = (__bf16)Bm[(size_t)k * N + ct * 16 + n];
  }
  __syncthreads();
  int lane = threadIdx.x & 31, wave = threadIdx.x >> 5;
  if (wave >= 4) return;  // 4 row tiles (M=64)
  int hi = lane >> 4, m = lane & 15, n = lane & 15;
  const float* Ar = A + (size_t)(wave * 16 + m) * 128;
  v8f c = {0.f, 0.f, 0.f, 0.f, 0.f, 0.f, 0.f, 0.f};
#pragma unroll
  for (int ks = 0; ks < 4; ks++) {
    v8fl lo = *(const v8fl*)(Ar + ks * 32 + 8 * hi);
    v8fl up = *(const v8fl*)(Ar + ks * 32 + 16 + 8 * hi);
    v16bf a;
#pragma unroll
    for (int e = 0; e < 8; e++) { a[e] = (__bf16)lo[e]; a[e + 8] = (__bf16)up[e]; }
    v16bf bb = afrag(&pan[n * 128 + ks * 32], hi);
    c = wmma_bf16(a, bb, c);
  }
  float bs = bias[ct * 16 + n];
#pragma unroll
  for (int j = 0; j < 8; j++) {
    int r = wave * 16 + j + 8 * hi;
    float v = c[j] + bs;
    if (act == 1) v = tanhf(v) * 5.f;
    else if (act == 2) v = tanhf(v);
    Cm[(size_t)r * N + ct * 16 + n] = v;
  }
}

// pack trans_w (12,128,256) -> Bt (128 x 3072), col = k*256+o
__global__ void k_pack(const float* __restrict__ tw, float* __restrict__ Bt) {
  int i = blockIdx.x * 256 + threadIdx.x;
  if (i >= NCP_ * 128 * 256) return;
  int k = i / (128 * 256);
  int rem = i - k * 128 * 256;
  int f = rem >> 8, o = rem & 255;
  Bt[(size_t)f * 3072 + k * 256 + o] = tw[i];
}

// =====================================================================
// TPS warp of prototypes + offset/scale + masked distance (mode 0) or
// output gather for the argmin prototype (mode 1).
// =====================================================================
__global__ void k_warp(const float* __restrict__ proto, const float* __restrict__ M28,
                       const float* __restrict__ ybase, const float* __restrict__ tmap,
                       const float* __restrict__ off, const float* __restrict__ scl,
                       const float* __restrict__ inp_s, const float* __restrict__ mask_s,
                       const int* __restrict__ sel, float* __restrict__ dist_out,
                       float* __restrict__ out, int mode) {
  int b = blockIdx.y;
  int p = mode ? sel[b] : blockIdx.x;
  __shared__ float ps[T_ * D_], is_[T_ * D_], msr[T_], tm[24], offs[D_], sef[D_], red[256];
  for (int i = threadIdx.x; i < T_ * D_; i += 256) {
    ps[i] = proto[(size_t)p * T_ * D_ + i];
    is_[i] = inp_s[(size_t)b * T_ * D_ + i];
  }
  for (int i = threadIdx.x; i < T_; i += 256) msr[i] = mask_s[b * T_ + i];
  if (threadIdx.x < 24)
    tm[threadIdx.x] = tmap[(size_t)b * 3072 + (threadIdx.x >> 1) * 256 + (threadIdx.x & 1) * 128 + p];
  if (threadIdx.x >= 32 && threadIdx.x < 32 + D_) {
    int d = threadIdx.x - 32;
    offs[d] = off[(size_t)b * (P_ * D_) + p * D_ + d];
    sef[d] = scl[(size_t)b * (P_ * D_) + p * D_ + d] * 0.75f + 1.25f;
  }
  __syncthreads();
  float acc = 0.f;
  for (int t = threadIdx.x; t < T_; t += 256) {
    const float* Mr = M28 + t * 28;
    float y = ybase[t];
#pragma unroll
    for (int i = 0; i < 24; i++) y += Mr[i + 2] * tm[i];
    float pos = (y + 1.f) * 0.5f * (float)(T_ - 1);
    pos = fminf(fmaxf(pos, 0.f), (float)(T_ - 1));
    int i0 = (int)floorf(pos);
    if (i0 > T_ - 1) i0 = T_ - 1;
    int i1 = min(i0 + 1, T_ - 1);
    float w = pos - (float)i0;
    float mw = msr[t];
    float s = 0.f;
#pragma unroll
    for (int d = 0; d < D_; d++) {
      float g = ps[i0 * D_ + d] * (1.f - w) + ps[i1 * D_ + d] * w;
      float val = (g + offs[d]) * sef[d];
      if (mode) {
        out[(size_t)b * T_ * D_ + t * D_ + d] = val;
      } else {
        float df = is_[t * D_ + d] - val;
        s += df * df;
      }
    }
    acc += mw * s;
  }
  if (!mode) {
    red[threadIdx.x] = acc;
    __syncthreads();
    for (int o = 128; o > 0; o >>= 1) {
      if (threadIdx.x < o) red[threadIdx.x] += red[threadIdx.x + o];
      __syncthreads();
    }
    if (threadIdx.x == 0) dist_out[b * P_ + p] = red[0];
  }
}

// argmin over prototypes (first-min like jnp.argmin) + int outputs as float
__global__ void k_argmin(const float* __restrict__ dist, const int* __restrict__ label,
                         float* __restrict__ idx_f, float* __restrict__ label_f,
                         int* __restrict__ idx_i) {
  int b = blockIdx.x;
  if (threadIdx.x == 0) {
    float best = dist[b * P_]; int bi = 0;
    for (int p = 1; p < P_; p++) {
      float v = dist[b * P_ + p];
      if (v < best) { best = v; bi = p; }
    }
    idx_i[b] = bi;
    idx_f[b] = (float)bi;
    label_f[b] = (float)label[b];
  }
}

// =====================================================================
extern "C" void kernel_launch(void* const* d_in, const int* in_sizes, int n_in,
                              void* d_out, int out_size, void* d_ws, size_t ws_size,
                              hipStream_t stream) {
  const float* input_seq = (const float*)d_in[0];
  const int*   label     = (const int*)d_in[1];
  const int*   mask      = (const int*)d_in[2];
  const float* proto     = (const float*)d_in[3];
  const float* conv1_w   = (const float*)d_in[4];
  const float* conv1_b   = (const float*)d_in[5];
  const float* bn1_g     = (const float*)d_in[6];
  const float* bn1_b     = (const float*)d_in[7];
  const float* conv2_w   = (const float*)d_in[8];
  const float* conv2_b   = (const float*)d_in[9];
  const float* bn2_g     = (const float*)d_in[10];
  const float* bn2_b     = (const float*)d_in[11];
  const float* conv3_w   = (const float*)d_in[12];
  const float* conv3_b   = (const float*)d_in[13];
  const float* bn3_g     = (const float*)d_in[14];
  const float* bn3_b     = (const float*)d_in[15];
  const float* final_w   = (const float*)d_in[16];
  const float* final_b   = (const float*)d_in[17];
  const float* trans_w   = (const float*)d_in[18];
  const float* trans_b   = (const float*)d_in[19];
  const float* offset_w  = (const float*)d_in[20];
  const float* offset_b  = (const float*)d_in[21];
  const float* scale_w   = (const float*)d_in[22];
  const float* scale_b   = (const float*)d_in[23];

  float* dout = (float*)d_out;
  float* out_out   = dout;                       // (B,T,D)  259840
  float* out_inps  = dout + 259840;              // (B,T,D)  259840
  float* out_dist  = dout + 519680;              // (B,P)    8192
  float* out_idx   = dout + 527872;              // (B)      64
  float* out_label = dout + 527936;              // (B)      64
  float* out_masks = dout + 528000;              // (B,T)    25984

  // ---- workspace layout (offsets in floats; all multiples of 64) ----
  float* ws = (float*)d_ws;
  size_t o = 0;
  auto alloc = [&](size_t nfloats) { float* p = ws + o; o += (nfloats + 63) & ~(size_t)63; return p; };
  float* wM28   = alloc(T_ * 28);
  float* wYb    = alloc(T_);
  float* wA1    = alloc(128);
  float* wB1    = alloc(128);
  float* wA2    = alloc(256);
  float* wB2    = alloc(256);
  float* wA3    = alloc(128);
  float* wB3    = alloc(128);
  float* wPool  = alloc(B_ * 128);
  float* wFeat  = alloc(B_ * 128);
  int*   wIdx   = (int*)alloc(64);
  float* wTmap  = alloc((size_t)B_ * 3072);
  float* wOff   = alloc((size_t)B_ * P_ * D_);
  float* wScl   = alloc((size_t)B_ * P_ * D_);
  float* wBt    = alloc((size_t)128 * 3072);
  float* wH1    = alloc((size_t)ROWS_ * 128);   // 13.3 MB, reused as h3
  float* wH2    = alloc((size_t)ROWS_ * 256);   // 26.6 MB
  __bf16* wXp   = (__bf16*)alloc(((size_t)B_ * RP1_ * 32 + 1) / 2);    // 1.7 MB
  __bf16* wAct1 = (__bf16*)alloc(((size_t)B_ * RP2_ * 128 + 1) / 2);   // 6.7 MB
  __bf16* wAct2 = (__bf16*)alloc(((size_t)B_ * RP3_ * 256 + 1) / 2);   // 13.4 MB
  float* wH3 = wH1;

  // independent constant / smoothing work
  hipLaunchKernelGGL(k_tps, dim3(1), dim3(64), 0, stream, wM28, wYb);
  hipLaunchKernelGGL(k_smooth, dim3(B_), dim3(256), 0, stream,
                     input_seq, mask, out_inps, out_masks);

  // encoder
  hipLaunchKernelGGL(k_actx, dim3((B_ * RP1_ * 32 + 255) / 256), dim3(256), 0, stream,
                     input_seq, wXp);
  hipLaunchKernelGGL(k_conv1, dim3(8, 64), dim3(256), 0, stream, wXp, conv1_w, conv1_b, wH1);
  hipLaunchKernelGGL(k_stats, dim3(128), dim3(256), 0, stream, wH1, 128, bn1_g, bn1_b, wA1, wB1);
  hipLaunchKernelGGL(k_act, dim3((B_ * RP2_ * 128 + 255) / 256), dim3(256), 0, stream,
                     wH1, wA1, wB1, wAct1, 128, 2, RP2_);
  hipLaunchKernelGGL(k_conv2, dim3(16, 64), dim3(256), 0, stream, wAct1, conv2_w, conv2_b, wH2);
  hipLaunchKernelGGL(k_stats, dim3(256), dim3(256), 0, stream, wH2, 256, bn2_g, bn2_b, wA2, wB2);
  hipLaunchKernelGGL(k_act, dim3((B_ * RP3_ * 256 + 255) / 256), dim3(256), 0, stream,
                     wH2, wA2, wB2, wAct2, 256, 1, RP3_);
  hipLaunchKernelGGL(k_conv3, dim3(8, 64), dim3(256), 0, stream, wAct2, conv3_w, conv3_b, wH3);
  hipLaunchKernelGGL(k_stats, dim3(128), dim3(256), 0, stream, wH3, 128, bn3_g, bn3_b, wA3, wB3);
  hipLaunchKernelGGL(k_pool, dim3(B_), dim3(128), 0, stream, wH3, wA3, wB3, wPool);
  hipLaunchKernelGGL(k_gemm128, dim3(8), dim3(256), 0, stream,
                     wPool, final_w, final_b, wFeat, 128, 0);

  // heads
  hipLaunchKernelGGL(k_pack, dim3((NCP_ * 128 * 256 + 255) / 256), dim3(256), 0, stream,
                     trans_w, wBt);
  hipLaunchKernelGGL(k_gemm128, dim3(192), dim3(256), 0, stream,
                     wFeat, wBt, trans_b, wTmap, 3072, 1);        // tanh*5
  hipLaunchKernelGGL(k_gemm128, dim3(80), dim3(256), 0, stream,
                     wFeat, offset_w, offset_b, wOff, 1280, 2);   // tanh
  hipLaunchKernelGGL(k_gemm128, dim3(80), dim3(256), 0, stream,
                     wFeat, scale_w, scale_b, wScl, 1280, 2);     // tanh

  // distances, argmin, final gather
  hipLaunchKernelGGL(k_warp, dim3(P_, B_), dim3(256), 0, stream,
                     proto, wM28, wYb, wTmap, wOff, wScl,
                     out_inps, out_masks, (const int*)nullptr, out_dist, out_out, 0);
  hipLaunchKernelGGL(k_argmin, dim3(B_), dim3(32), 0, stream,
                     out_dist, label, out_idx, out_label, wIdx);
  hipLaunchKernelGGL(k_warp, dim3(1, B_), dim3(256), 0, stream,
                     proto, wM28, wYb, wTmap, wOff, wScl,
                     out_inps, out_masks, wIdx, out_dist, out_out, 1);

  (void)in_sizes; (void)n_in; (void)out_size; (void)ws_size;
}
